// MultiHeadCausalAttention_11982958756144
// MI455X (gfx1250) — matrix operand order repaired and checked by reading, not compiled, not run on hardware
//
#include <hip/hip_runtime.h>
#include <hip/hip_bf16.h>
#include <math.h>

// ---------------------------------------------------------------------------
// Shapes (fixed by the reference)
// ---------------------------------------------------------------------------
#define BB   2
#define TT   2048
#define DD   1024
#define HH   16
#define DHD  64
#define MT   (BB * TT)        // 4096 rows of activations
#define DFF  2730
#define DFFP 2752             // padded to a multiple of 64
#define EPSV 1e-5f

// ---------------------------------------------------------------------------
// WMMA types (CDNA5 / gfx1250, wave32)
// ---------------------------------------------------------------------------
typedef __attribute__((ext_vector_type(16))) __bf16 v16bf;
typedef __attribute__((ext_vector_type(8)))  float  v8f;

union FragBF {
  v16bf v;
  unsigned int u[8];
  unsigned short s[16];
};

__device__ __forceinline__ v8f wmma_bf16(v16bf a, v16bf b, v8f c) {
  // D = A(16x32 bf16) * B(32x16 bf16) + C(16x16 f32)
  return __builtin_amdgcn_wmma_f32_16x16x32_bf16(
      /*neg_a=*/false, a, /*neg_b=*/false, b,
      /*c_mod=*/(short)0, c, /*reuse_a=*/false, /*reuse_b=*/false);
}

// float -> bf16, round-to-nearest-even
__device__ __forceinline__ unsigned short f2bf(float f) {
  unsigned int u = __float_as_uint(f);
  unsigned int r = u + 0x7FFFu + ((u >> 16) & 1u);
  if ((u & 0x7F800000u) == 0x7F800000u) r = u;   // inf/NaN: truncate
  return (unsigned short)(r >> 16);
}
__device__ __forceinline__ float bf2f(unsigned short s) {
  return __uint_as_float(((unsigned int)s) << 16);
}

// A-fragment loader (16x32 bf16, M x K), row-major tile with row stride `ld`
// Layout used here (contiguous K): lane M = lane&15; VGPR j holds K pair:
//   K = (lane>=16 ? 16:0) + j*2 + {0,1}
__device__ __forceinline__ v16bf load_a_frag(const unsigned short* tile, int ld, int lane) {
  const int m   = lane & 15;
  const int hi16 = (lane & 16) ? 16 : 0;
  const unsigned short* row = tile + (size_t)m * ld;
  FragBF f;
#pragma unroll
  for (int j = 0; j < 8; ++j) {
    const int k = hi16 + j * 2;
    f.u[j] = *(const unsigned int*)(row + k);
  }
  return f.v;
}

// Async global->LDS copy of 64 contiguous bytes (4x B128), ASYNCcnt-tracked.
// LDS byte address = low 32 bits of the flat shared pointer (ISA 10.2:
// LDS aperture address is addr[31:0]); instruction offset applies to both
// the LDS and the global address.
__device__ __forceinline__ void async_copy64(unsigned short* lds_dst, const unsigned short* gsrc) {
  const unsigned lds = (unsigned)(size_t)lds_dst;
  const unsigned long long ga = (unsigned long long)(size_t)gsrc;
  asm volatile(
      "global_load_async_to_lds_b128 %0, %1, off\n\t"
      "global_load_async_to_lds_b128 %0, %1, off offset:16\n\t"
      "global_load_async_to_lds_b128 %0, %1, off offset:32\n\t"
      "global_load_async_to_lds_b128 %0, %1, off offset:48"
      :: "v"(lds), "v"(ga) : "memory");
}
__device__ __forceinline__ void wait_async0() {
  asm volatile("s_wait_asynccnt 0x0" ::: "memory");
}

// ---------------------------------------------------------------------------
// float->bf16 convert with zero padding: src[R x C] -> dst[RP x CP]
// ---------------------------------------------------------------------------
__global__ void convert_pad_bf16(const float* __restrict__ src, unsigned short* __restrict__ dst,
                                 int R, int C, int RP, int CP) {
  size_t i = (size_t)blockIdx.x * blockDim.x + threadIdx.x;
  size_t n = (size_t)RP * CP;
  if (i >= n) return;
  int r = (int)(i / CP), c = (int)(i % CP);
  dst[i] = (r < R && c < C) ? f2bf(src[(size_t)r * C + c]) : (unsigned short)0;
}

// ---------------------------------------------------------------------------
// RMSNorm: one block per row of DD=1024, output bf16
// ---------------------------------------------------------------------------
__global__ void rmsnorm_kernel(const float* __restrict__ X, const float* __restrict__ g,
                               unsigned short* __restrict__ Hout) {
  __shared__ float red[256];
  const size_t row = blockIdx.x;
  const float* xr = X + row * DD;
  float ss = 0.f;
  for (int c = threadIdx.x; c < DD; c += 256) { float v = xr[c]; ss += v * v; }
  red[threadIdx.x] = ss;
  __syncthreads();
  for (int s = 128; s > 0; s >>= 1) {
    if (threadIdx.x < s) red[threadIdx.x] += red[threadIdx.x + s];
    __syncthreads();
  }
  const float inv = rsqrtf(red[0] / (float)DD + EPSV);
  for (int c = threadIdx.x; c < DD; c += 256)
    Hout[row * DD + c] = f2bf(g[c] * xr[c] * inv);
}

// ---------------------------------------------------------------------------
// bf16 WMMA GEMM: C[M,N] = A[M,K](bf16,rm) * B[K,N](bf16,rm)
// Block tile 128(M) x 64(N), K step 64, 8 waves (4x2), wave tile 32x32,
// 8 WMMAs per wave per k-step.  Double-buffered software pipeline:
//   - A tile t+1 prefetched by async global->LDS DMA (ASYNCcnt)
//   - B tile t+1 prefetched into registers, transposed into LDS after compute
//   - single barrier per k-step
// mode 0: store f32; mode 1: store bf16; mode 2: store f32 + residual R
// M%128==0, N%64==0, K%64==0 (guaranteed by padding).
// ---------------------------------------------------------------------------
__global__ __launch_bounds__(256)
void gemm_bf16(const unsigned short* __restrict__ A, const unsigned short* __restrict__ B,
               void* __restrict__ Cout, const float* __restrict__ R,
               int M, int N, int K, int mode) {
  __shared__ unsigned short As[2][128 * 64];   // 2 x 16 KB [m][k]
  __shared__ unsigned short Bs[2][64 * 64];    // 2 x  8 KB [n][k]  (transposed)

  const int tid  = threadIdx.x;
  const int lane = tid & 31;
  const int w    = tid >> 5;
  const int wm   = w & 3;        // 0..3 -> M sub-tile
  const int wn   = w >> 2;       // 0..1 -> N sub-tile
  const int m0   = blockIdx.y * 128;
  const int n0   = blockIdx.x * 64;
  const int hi16 = (lane & 16) ? 16 : 0;
  const int hi8  = (lane & 16) ? 8 : 0;
  const int ln   = lane & 15;

  // staging coordinates
  const int arow = tid >> 1, aseg = tid & 1;       // A: 2 threads/row, 32 bf16 each
  const int bk   = tid >> 2, bnc = (tid & 3) * 16; // B: 4 threads/row, 16 bf16 each

  const unsigned short* Arow = A + (size_t)(m0 + arow) * K + aseg * 32;
  const unsigned short* Brow = B + (size_t)bk * N + n0 + bnc;

  v8f acc[2][2] = {};
  const int nk = K >> 6;

  // ---- prologue: stage tile 0
  async_copy64(&As[0][arow * 64 + aseg * 32], Arow);
  {
    union { uint4 q[2]; unsigned short s[16]; } u;
    u.q[0] = ((const uint4*)Brow)[0];
    u.q[1] = ((const uint4*)Brow)[1];
#pragma unroll
    for (int i = 0; i < 16; ++i) Bs[0][(bnc + i) * 64 + bk] = u.s[i];
  }
  wait_async0();
  __syncthreads();

  for (int t = 0; t < nk; ++t) {
    const int cur = t & 1, nxt = cur ^ 1;
    const bool has_next = (t + 1 < nk);

    // ---- issue prefetch of tile t+1 (A via async DMA, B into registers)
    uint4 pb0{}, pb1{};
    if (has_next) {
      async_copy64(&As[nxt][arow * 64 + aseg * 32], Arow + (size_t)(t + 1) * 64);
      const unsigned short* src = Brow + (size_t)(t + 1) * 64 * N;
      pb0 = ((const uint4*)src)[0];
      pb1 = ((const uint4*)src)[1];
    }

    // ---- compute on tile t (8 WMMAs) while DMA + loads are in flight
    v16bf a[2][2], bq[2][2];
#pragma unroll
    for (int kc = 0; kc < 2; ++kc) {
      a[0][kc]  = load_a_frag(&As[cur][(wm * 32 + 0)  * 64 + kc * 32], 64, lane);
      a[1][kc]  = load_a_frag(&As[cur][(wm * 32 + 16) * 64 + kc * 32], 64, lane);
      bq[0][kc] = *(const v16bf*)(&Bs[cur][(wn * 32 + 0  + ln) * 64 + kc * 32 + hi16]);
      bq[1][kc] = *(const v16bf*)(&Bs[cur][(wn * 32 + 16 + ln) * 64 + kc * 32 + hi16]);
    }
#pragma unroll
    for (int kc = 0; kc < 2; ++kc) {
      acc[0][0] = wmma_bf16(a[0][kc], bq[0][kc], acc[0][0]);
      acc[0][1] = wmma_bf16(a[0][kc], bq[1][kc], acc[0][1]);
      acc[1][0] = wmma_bf16(a[1][kc], bq[0][kc], acc[1][0]);
      acc[1][1] = wmma_bf16(a[1][kc], bq[1][kc], acc[1][1]);
    }

    // ---- finish staging tile t+1, then one barrier
    if (has_next) {
      union { uint4 q[2]; unsigned short s[16]; } u;
      u.q[0] = pb0;
      u.q[1] = pb1;
#pragma unroll
      for (int i = 0; i < 16; ++i) Bs[nxt][(bnc + i) * 64 + bk] = u.s[i];
      wait_async0();          // A tile t+1 resident
    }
    __syncthreads();
  }

  // --- epilogue: C layout row = mi*16 + hi8 + i, col = ni*16 + ln
#pragma unroll
  for (int mi = 0; mi < 2; ++mi)
#pragma unroll
    for (int ni = 0; ni < 2; ++ni)
#pragma unroll
      for (int i = 0; i < 8; ++i) {
        const size_t row = (size_t)m0 + wm * 32 + mi * 16 + hi8 + i;
        const size_t col = (size_t)n0 + wn * 32 + ni * 16 + ln;
        const float v = acc[mi][ni][i];
        if (mode == 1) {
          ((unsigned short*)Cout)[row * N + col] = f2bf(v);
        } else {
          const float r = (mode == 2) ? R[row * N + col] : 0.f;
          ((float*)Cout)[row * N + col] = r + v;
        }
      }
}

// ---------------------------------------------------------------------------
// Flash attention, causal.  NOTE: the reference's ALiBi bias is
// -slope * clamp(j - i, min=0), which is identically ZERO wherever the causal
// mask (j <= i) keeps the score; so only causal masking is required.
// One wave per (b, h, 16-query block). K step = 32 keys.
// ---------------------------------------------------------------------------
__global__ __launch_bounds__(32)
void attn_kernel(const unsigned short* __restrict__ Qb, const unsigned short* __restrict__ Kb,
                 const unsigned short* __restrict__ Vb, unsigned short* __restrict__ Ctx) {
  __shared__ unsigned short ps[16 * 32];   // P tile  [qrow][key]
  __shared__ unsigned short vt[64 * 32];   // V^T tile [dh][key]

  const int lane = threadIdx.x;
  const int ln   = lane & 15;
  const int hi   = (lane & 16) ? 1 : 0;
  const int hi16 = hi * 16;
  const int hi8  = hi * 8;
  const int q0   = blockIdx.x * 16;
  const int h    = blockIdx.y;
  const int b    = blockIdx.z;
  const size_t rowbase = (size_t)b * TT;
  const size_t hoff    = (size_t)h * DHD;

  // Q fragments (16 x 64, two 32-wide K chunks), straight from global
  v16bf aq0 = load_a_frag(Qb + (rowbase + q0) * (HH * DHD) + hoff,      HH * DHD, lane);
  v16bf aq1 = load_a_frag(Qb + (rowbase + q0) * (HH * DHD) + hoff + 32, HH * DHD, lane);

  v8f o[4] = {};
  float mrow[8], lrow[8];
#pragma unroll
  for (int i = 0; i < 8; ++i) { mrow[i] = -1e30f; lrow[i] = 0.f; }

  const int ksteps = (q0 + 16 + 31) / 32;   // causal coverage, 32 keys/step
  for (int kt = 0; kt < ksteps; ++kt) {
    const int kb0 = kt * 32;

    // ---- S = Q K^T (two 16-key column tiles)
    v8f s[2] = {};
#pragma unroll
    for (int ni = 0; ni < 2; ++ni) {
      const size_t krow = rowbase + kb0 + ni * 16 + ln;
      v16bf bk0 = *(const v16bf*)(Kb + krow * (HH * DHD) + hoff + 0  + hi16);
      v16bf bk1 = *(const v16bf*)(Kb + krow * (HH * DHD) + hoff + 32 + hi16);
      s[ni] = wmma_bf16(aq0, bk0, s[ni]);
      s[ni] = wmma_bf16(aq1, bk1, s[ni]);
    }

    // ---- scale + causal mask + online softmax
    float tmax[8];
#pragma unroll
    for (int i = 0; i < 8; ++i) {
      const int qr = q0 + hi8 + i;
#pragma unroll
      for (int ni = 0; ni < 2; ++ni) {
        const int kc = kb0 + ni * 16 + ln;
        float v = s[ni][i] * 0.125f;        // DH^-0.5 = 1/8
        if (kc > qr) v = -INFINITY;
        s[ni][i] = v;
      }
      tmax[i] = fmaxf(s[0][i], s[1][i]);
    }
#pragma unroll
    for (int off = 1; off < 16; off <<= 1)  // reduce across 16-lane half
#pragma unroll
      for (int i = 0; i < 8; ++i)
        tmax[i] = fmaxf(tmax[i], __shfl_xor(tmax[i], off, 32));

#pragma unroll
    for (int i = 0; i < 8; ++i) {
      const float mn    = fmaxf(mrow[i], tmax[i]);
      const float alpha = __expf(mrow[i] - mn);
      const float p0    = __expf(s[0][i] - mn);
      const float p1    = __expf(s[1][i] - mn);
      float rs = p0 + p1;
#pragma unroll
      for (int off = 1; off < 16; off <<= 1) rs += __shfl_xor(rs, off, 32);
      lrow[i] = lrow[i] * alpha + rs;
      mrow[i] = mn;
#pragma unroll
      for (int nio = 0; nio < 4; ++nio) o[nio][i] *= alpha;
      ps[(hi8 + i) * 32 + ln]      = f2bf(p0);
      ps[(hi8 + i) * 32 + 16 + ln] = f2bf(p1);
    }

    // ---- stage V^T: vector-load one key row (64 dh) per lane, transpose on
    //      the LDS-write side
    {
      const unsigned short* vr = Vb + (rowbase + kb0 + lane) * (HH * DHD) + hoff;
      union { uint4 q; unsigned short s[8]; } u;
#pragma unroll
      for (int c = 0; c < 8; ++c) {
        u.q = ((const uint4*)vr)[c];
#pragma unroll
        for (int i = 0; i < 8; ++i) vt[(c * 8 + i) * 32 + lane] = u.s[i];
      }
    }
    __syncthreads();

    // ---- O += P V
    v16bf ap = load_a_frag(ps, 32, lane);
#pragma unroll
    for (int nio = 0; nio < 4; ++nio) {
      v16bf bv = *(const v16bf*)(vt + (nio * 16 + ln) * 32 + hi16);
      o[nio] = wmma_bf16(ap, bv, o[nio]);
    }
    __syncthreads();   // protect ps/vt before next iteration overwrites
  }

  // ---- normalize and write ctx (bf16)
#pragma unroll
  for (int i = 0; i < 8; ++i) {
    const float inv = 1.0f / lrow[i];
    const size_t row = rowbase + q0 + hi8 + i;
#pragma unroll
    for (int nio = 0; nio < 4; ++nio)
      Ctx[row * (HH * DHD) + hoff + nio * 16 + ln] = f2bf(o[nio][i] * inv);
  }
}

// ---------------------------------------------------------------------------
// SwiGLU gate: g = bf16( silu(u1) * u2 ), elementwise over MT x DFFP
// ---------------------------------------------------------------------------
__global__ void gate_kernel(const unsigned short* __restrict__ U1,
                            const unsigned short* __restrict__ U2,
                            unsigned short* __restrict__ G, size_t n) {
  size_t i = (size_t)blockIdx.x * blockDim.x + threadIdx.x;
  if (i >= n) return;
  const float a = bf2f(U1[i]);
  const float s = a / (1.0f + __expf(-a));
  G[i] = f2bf(s * bf2f(U2[i]));
}

// ---------------------------------------------------------------------------
// Host orchestration
// ---------------------------------------------------------------------------
extern "C" void kernel_launch(void* const* d_in, const int* in_sizes, int n_in,
                              void* d_out, int out_size, void* d_ws, size_t ws_size,
                              hipStream_t stream) {
  const float* x  = (const float*)d_in[0];
  const float* g1 = (const float*)d_in[1];
  const float* Wq = (const float*)d_in[2];
  const float* Wk = (const float*)d_in[3];
  const float* Wv = (const float*)d_in[4];
  const float* Wo = (const float*)d_in[5];
  const float* g2 = (const float*)d_in[6];
  const float* W1 = (const float*)d_in[7];
  const float* W2 = (const float*)d_in[8];
  const float* W3 = (const float*)d_in[9];
  float* out = (float*)d_out;

  // Workspace bump allocator (total ~155 MB)
  char* ws = (char*)d_ws;
  size_t off = 0;
  auto alloc = [&](size_t bytes) -> void* {
    void* p = ws + off;
    off = (off + bytes + 255) & ~(size_t)255;
    return p;
  };
  unsigned short* hb  = (unsigned short*)alloc((size_t)MT * DD * 2);
  unsigned short* wqb = (unsigned short*)alloc((size_t)DD * DD * 2);
  unsigned short* wkb = (unsigned short*)alloc((size_t)DD * DD * 2);
  unsigned short* wvb = (unsigned short*)alloc((size_t)DD * DD * 2);
  unsigned short* wob = (unsigned short*)alloc((size_t)DD * DD * 2);
  unsigned short* w1b = (unsigned short*)alloc((size_t)DD * DFFP * 2);
  unsigned short* w2b = (unsigned short*)alloc((size_t)DD * DFFP * 2);
  unsigned short* w3b = (unsigned short*)alloc((size_t)DFFP * DD * 2);
  unsigned short* qb  = (unsigned short*)alloc((size_t)MT * DD * 2);
  unsigned short* kb  = (unsigned short*)alloc((size_t)MT * DD * 2);
  unsigned short* vb  = (unsigned short*)alloc((size_t)MT * DD * 2);
  unsigned short* ctb = (unsigned short*)alloc((size_t)MT * DD * 2);
  float*          x1f = (float*)        alloc((size_t)MT * DD * 4);
  unsigned short* h2b = (unsigned short*)alloc((size_t)MT * DD * 2);
  unsigned short* u1b = (unsigned short*)alloc((size_t)MT * DFFP * 2);
  unsigned short* u2b = (unsigned short*)alloc((size_t)MT * DFFP * 2);
  unsigned short* gb  = (unsigned short*)alloc((size_t)MT * DFFP * 2);

  auto cvt = [&](const float* s, unsigned short* d, int R, int C, int RP, int CP) {
    size_t n = (size_t)RP * CP;
    convert_pad_bf16<<<dim3((unsigned)((n + 255) / 256)), dim3(256), 0, stream>>>(s, d, R, C, RP, CP);
  };

  // 1) weights -> bf16 (pad DFF 2730 -> 2752 with zeros)
  cvt(Wq, wqb, DD, DD, DD, DD);
  cvt(Wk, wkb, DD, DD, DD, DD);
  cvt(Wv, wvb, DD, DD, DD, DD);
  cvt(Wo, wob, DD, DD, DD, DD);
  cvt(W1, w1b, DD, DFF, DD, DFFP);
  cvt(W2, w2b, DD, DFF, DD, DFFP);
  cvt(W3, w3b, DFF, DD, DFFP, DD);

  // 2) h = rmsnorm(x, g1)  (bf16)
  rmsnorm_kernel<<<dim3(MT), dim3(256), 0, stream>>>(x, g1, hb);

  // 3) q,k,v = h @ W{q,k,v}  (bf16 out)
  const dim3 gemm_blk(256);
  const dim3 grid_qkv(DD / 64, MT / 128);
  gemm_bf16<<<grid_qkv, gemm_blk, 0, stream>>>(hb, wqb, qb, nullptr, MT, DD, DD, 1);
  gemm_bf16<<<grid_qkv, gemm_blk, 0, stream>>>(hb, wkb, kb, nullptr, MT, DD, DD, 1);
  gemm_bf16<<<grid_qkv, gemm_blk, 0, stream>>>(hb, wvb, vb, nullptr, MT, DD, DD, 1);

  // 4) ctx = causal flash attention (ALiBi bias is zero on causal region)
  attn_kernel<<<dim3(TT / 16, HH, BB), dim3(32), 0, stream>>>(qb, kb, vb, ctb);

  // 5) x1 = x + ctx @ Wo  (f32)
  gemm_bf16<<<grid_qkv, gemm_blk, 0, stream>>>(ctb, wob, x1f, x, MT, DD, DD, 2);

  // 6) h2 = rmsnorm(x1, g2)  (bf16)
  rmsnorm_kernel<<<dim3(MT), dim3(256), 0, stream>>>(x1f, g2, h2b);

  // 7) u1 = h2 @ W1, u2 = h2 @ W2  (bf16 out, padded N)
  const dim3 grid_ff(DFFP / 64, MT / 128);
  gemm_bf16<<<grid_ff, gemm_blk, 0, stream>>>(h2b, w1b, u1b, nullptr, MT, DFFP, DD, 1);
  gemm_bf16<<<grid_ff, gemm_blk, 0, stream>>>(h2b, w2b, u2b, nullptr, MT, DFFP, DD, 1);

  // 8) g = silu(u1) * u2   (pad region stays zero -> no effect on W3 contraction)
  {
    size_t n = (size_t)MT * DFFP;
    gate_kernel<<<dim3((unsigned)((n + 255) / 256)), dim3(256), 0, stream>>>(u1b, u2b, gb, n);
  }

  // 9) out = x1 + g @ W3  (f32, straight into d_out)
  gemm_bf16<<<dim3(DD / 64, MT / 128), gemm_blk, 0, stream>>>(gb, w3b, out, x1f, MT, DD, DFFP, 2);
}
